// PointNetPPEncoder_38611755991865
// MI455X (gfx1250) — compile-verified, matrix operand order checked by reading, not run on hardware
//
#include <hip/hip_runtime.h>
#include <hip/hip_fp16.h>

typedef __attribute__((ext_vector_type(16))) _Float16 v16h;
typedef __attribute__((ext_vector_type(8)))  float    v8f;

#define CDIV(a,b) (((a)+(b)-1)/(b))
#define RUP32(x) (((x)+31)&~31)

// ---------------------------------------------------------------------------
// 1) Transpose input (B,6,N) -> xyz_t (B,N,3) and pts_t (B,N,6)
// ---------------------------------------------------------------------------
__global__ void k_transpose_in(const float* __restrict__ xyz,
                               float* __restrict__ xyz_t,
                               float* __restrict__ pts_t, int N) {
  int idx = blockIdx.x * blockDim.x + threadIdx.x;
  if (idx >= 8 * N) return;
  int b = idx / N, n = idx % N;
  for (int c = 0; c < 6; c++) {
    float v = xyz[((size_t)b * 6 + c) * N + n];
    pts_t[((size_t)b * N + n) * 6 + c] = v;
    if (c < 3) xyz_t[((size_t)b * N + n) * 3 + c] = v;
  }
}

// ---------------------------------------------------------------------------
// 2) Farthest point sampling: one block per batch, sequential S iterations
//    with block-wide argmax (first-occurrence tie-break, like jnp.argmax).
// ---------------------------------------------------------------------------
__global__ void k_fps(const float* __restrict__ xyz_t, int N, int S,
                      float* __restrict__ dist, int* __restrict__ fidx,
                      float* __restrict__ new_xyz) {
  int b = blockIdx.x;
  const float* X = xyz_t + (size_t)b * N * 3;
  float* D = dist + (size_t)b * N;
  __shared__ float s_val[256];
  __shared__ int   s_idx[256];
  __shared__ float s_c[3];
  __shared__ int   s_far;
  int t = threadIdx.x;
  for (int n = t; n < N; n += 256) D[n] = 1e10f;
  if (t == 0) s_far = 0;
  __syncthreads();
  for (int i = 0; i < S; i++) {
    if (t == 0) {
      int far = s_far;
      fidx[b * S + i] = far;
      float cx = X[far * 3 + 0], cy = X[far * 3 + 1], cz = X[far * 3 + 2];
      new_xyz[((size_t)b * S + i) * 3 + 0] = cx;
      new_xyz[((size_t)b * S + i) * 3 + 1] = cy;
      new_xyz[((size_t)b * S + i) * 3 + 2] = cz;
      s_c[0] = cx; s_c[1] = cy; s_c[2] = cz;
    }
    __syncthreads();
    float cx = s_c[0], cy = s_c[1], cz = s_c[2];
    float best = -1.0f; int bi = N;
    for (int n = t; n < N; n += 256) {
      float dx = X[n * 3] - cx, dy = X[n * 3 + 1] - cy, dz = X[n * 3 + 2] - cz;
      float d = dx * dx + dy * dy + dz * dz;
      float dn = fminf(D[n], d);
      D[n] = dn;
      if (dn > best) { best = dn; bi = n; }
    }
    s_val[t] = best; s_idx[t] = bi;
    __syncthreads();
    for (int off = 128; off > 0; off >>= 1) {
      if (t < off) {
        float ov = s_val[t + off]; int oi = s_idx[t + off];
        if (ov > s_val[t] || (ov == s_val[t] && oi < s_idx[t])) {
          s_val[t] = ov; s_idx[t] = oi;
        }
      }
      __syncthreads();
    }
    if (t == 0) s_far = s_idx[0];
    __syncthreads();
  }
}

// ---------------------------------------------------------------------------
// 3) Ball query: reference keeps the K smallest indices with sq <= r^2,
//    padding with the first hit -> ordered scan with early exit.
// ---------------------------------------------------------------------------
__global__ void k_ballquery(const float* __restrict__ xyz_t,
                            const float* __restrict__ new_xyz,
                            int N, int S, float r2, int K,
                            int* __restrict__ gidx) {
  int idx = blockIdx.x * blockDim.x + threadIdx.x;
  if (idx >= 8 * S) return;
  int b = idx / S, s = idx % S;
  const float* X = xyz_t + (size_t)b * N * 3;
  float cx = new_xyz[((size_t)b * S + s) * 3 + 0];
  float cy = new_xyz[((size_t)b * S + s) * 3 + 1];
  float cz = new_xyz[((size_t)b * S + s) * 3 + 2];
  int* out = gidx + ((size_t)b * S + s) * K;
  int cnt = 0, first = -1;
  for (int n = 0; n < N && cnt < K; n++) {
    float dx = X[n * 3] - cx, dy = X[n * 3 + 1] - cy, dz = X[n * 3 + 2] - cz;
    float d = dx * dx + dy * dy + dz * dz;
    if (d <= r2) { if (first < 0) first = n; out[cnt++] = n; }
  }
  if (first < 0) first = 0;
  for (; cnt < K; cnt++) out[cnt] = first;
}

// ---------------------------------------------------------------------------
// 4) Group: build X (M=B*S*K rows, Kp cols) f16 = [pts(gidx) | rel_xyz | 0pad]
// ---------------------------------------------------------------------------
__global__ void k_group(const float* __restrict__ pts_t,
                        const float* __restrict__ xyz_t,
                        const float* __restrict__ new_xyz,
                        const int* __restrict__ gidx,
                        int N, int S, int K, int Cin, int Kp,
                        _Float16* __restrict__ Xh) {
  int row = blockIdx.x * blockDim.x + threadIdx.x;
  int M = 8 * S * K;
  if (row >= M) return;
  int k = row % K, s = (row / K) % S, b = row / (S * K);
  int g = gidx[((size_t)b * S + s) * K + k];
  const float* P = pts_t + ((size_t)b * N + g) * Cin;
  _Float16* o = Xh + (size_t)row * Kp;
  for (int c = 0; c < Cin; c++) o[c] = (_Float16)P[c];
  const float* Xg = xyz_t + ((size_t)b * N + g) * 3;
  const float* C = new_xyz + ((size_t)b * S + s) * 3;
  o[Cin + 0] = (_Float16)(Xg[0] - C[0]);
  o[Cin + 1] = (_Float16)(Xg[1] - C[1]);
  o[Cin + 2] = (_Float16)(Xg[2] - C[2]);
  for (int c = Cin + 3; c < Kp; c++) o[c] = (_Float16)0.0f;
}

// ---------------------------------------------------------------------------
// 5) Weight convert: W (Cout, Creal) f32 -> Wp (Cpad, Kp) f16, zero padded
// ---------------------------------------------------------------------------
__global__ void k_wconv(const float* __restrict__ W, int Cout, int Creal,
                        int Cpad, int Kp, _Float16* __restrict__ Wp) {
  int idx = blockIdx.x * blockDim.x + threadIdx.x;
  if (idx >= Cpad * Kp) return;
  int e = idx / Kp, d = idx % Kp;
  Wp[idx] = (e < Cout && d < Creal) ? (_Float16)W[(size_t)e * Creal + d]
                                    : (_Float16)0.0f;
}

// ---------------------------------------------------------------------------
// 6) WMMA GEMM: Y[m][n] = sum_d A[m][d]*W[n][d] + bias[n], fp32 accum.
//    One wave computes a 32x32 output tile as 2x2 WMMA tiles: each A fragment
//    pair feeds two N-tiles and each B fragment pair feeds two M-tiles
//    (2 b128 loads per v_wmma instead of 4). K loop steps of 32 with
//    v_wmma_f32_16x16x32_f16; WGP-scope global_prefetch for next A K-chunk
//    (locality=3 -> scope WGP: pulls into all cache levels, unlike SYS).
//    Fragment layout (ISA 7.12.2, 16-bit A 16x32): lane r=lane&15 -> row,
//    hi=lane>>4 selects K-halves {8*hi..8*hi+7, 16+8*hi..23+8*hi}.
// ---------------------------------------------------------------------------
__global__ void k_gemm_wmma(const _Float16* __restrict__ A,
                            const _Float16* __restrict__ Wp,
                            const float* __restrict__ bias,
                            int M, int Cpad, int Kp, int Cout,
                            float* __restrict__ Y) {
  int wave = (blockIdx.x * blockDim.x + threadIdx.x) >> 5;
  int lane = threadIdx.x & 31;
  int Mt = M >> 5, Nt = Cpad >> 5;           // 32x32 tiles
  if (wave >= Mt * Nt) return;
  int mt = wave / Nt, nt = wave % Nt;
  int r = lane & 15, hi = lane >> 4;
  const _Float16* a0 = A  + (size_t)(mt * 32 + r) * Kp;
  const _Float16* a1 = a0 + (size_t)16 * Kp;
  const _Float16* b0 = Wp + (size_t)(nt * 32 + r) * Kp;
  const _Float16* b1 = b0 + (size_t)16 * Kp;
  v8f acc00 = {}, acc01 = {}, acc10 = {}, acc11 = {};
  union U { uint4 u[2]; v16h h; };
  for (int kk = 0; kk < Kp; kk += 32) {
    if (kk + 32 < Kp) {                      // hide next A chunk (streaming op)
      __builtin_prefetch(a0 + kk + 32, 0, 3);
      __builtin_prefetch(a1 + kk + 32, 0, 3);
    }
    U fa0, fa1, fb0, fb1;
    const uint4* pa0 = (const uint4*)(a0 + kk);
    const uint4* pa1 = (const uint4*)(a1 + kk);
    const uint4* pb0 = (const uint4*)(b0 + kk);
    const uint4* pb1 = (const uint4*)(b1 + kk);
    fa0.u[0] = pa0[hi]; fa0.u[1] = pa0[2 + hi];
    fa1.u[0] = pa1[hi]; fa1.u[1] = pa1[2 + hi];
    fb0.u[0] = pb0[hi]; fb0.u[1] = pb0[2 + hi];
    fb1.u[0] = pb1[hi]; fb1.u[1] = pb1[2 + hi];
    acc00 = __builtin_amdgcn_wmma_f32_16x16x32_f16(
        false, fa0.h, false, fb0.h, (short)0, acc00, false, false);
    acc01 = __builtin_amdgcn_wmma_f32_16x16x32_f16(
        false, fa0.h, false, fb1.h, (short)0, acc01, false, false);
    acc10 = __builtin_amdgcn_wmma_f32_16x16x32_f16(
        false, fa1.h, false, fb0.h, (short)0, acc10, false, false);
    acc11 = __builtin_amdgcn_wmma_f32_16x16x32_f16(
        false, fa1.h, false, fb1.h, (short)0, acc11, false, false);
  }
  // D layout: VGPR i -> M = i + 8*hi, N = lane&15 (per 16x16 sub-tile)
  int n0 = nt * 32 + r;
  int n1 = n0 + 16;
  float bv0 = (n0 < Cout) ? bias[n0] : 0.0f;
  float bv1 = (n1 < Cout) ? bias[n1] : 0.0f;
  float* y00 = Y + (size_t)(mt * 32) * Cpad + n0;
  float* y01 = Y + (size_t)(mt * 32) * Cpad + n1;
  float* y10 = Y + (size_t)(mt * 32 + 16) * Cpad + n0;
  float* y11 = Y + (size_t)(mt * 32 + 16) * Cpad + n1;
#pragma unroll
  for (int i = 0; i < 8; i++) {
    size_t ro = (size_t)(i + 8 * hi) * Cpad;
    y00[ro] = acc00[i] + bv0;
    y01[ro] = acc01[i] + bv1;
    y10[ro] = acc10[i] + bv0;
    y11[ro] = acc11[i] + bv1;
  }
}

// ---------------------------------------------------------------------------
// 7) BN batch stats: one block per channel, reduce sum/sumsq over M rows
// ---------------------------------------------------------------------------
__global__ void k_stats(const float* __restrict__ Y, int M, int Cpad,
                        float* __restrict__ meanvar) {
  int e = blockIdx.x;
  __shared__ float s1[256], s2[256];
  float a = 0.f, q = 0.f;
  for (int m = threadIdx.x; m < M; m += 256) {
    float v = Y[(size_t)m * Cpad + e];
    a += v; q += v * v;
  }
  s1[threadIdx.x] = a; s2[threadIdx.x] = q;
  __syncthreads();
  for (int off = 128; off > 0; off >>= 1) {
    if (threadIdx.x < off) {
      s1[threadIdx.x] += s1[threadIdx.x + off];
      s2[threadIdx.x] += s2[threadIdx.x + off];
    }
    __syncthreads();
  }
  if (threadIdx.x == 0) {
    float mean = s1[0] / (float)M;
    float var  = s2[0] / (float)M - mean * mean;
    meanvar[e] = mean;
    meanvar[Cpad + e] = var;
  }
}

// ---------------------------------------------------------------------------
// 8) BN + ReLU -> f16 activation (pad channels forced to 0 for next GEMM)
// ---------------------------------------------------------------------------
__global__ void k_bnrelu(const float* __restrict__ Y,
                         const float* __restrict__ meanvar,
                         const float* __restrict__ gamma,
                         const float* __restrict__ beta,
                         int M, int Cpad, int Cout,
                         _Float16* __restrict__ out) {
  size_t idx = (size_t)blockIdx.x * blockDim.x + threadIdx.x;
  if (idx >= (size_t)M * Cpad) return;
  int e = (int)(idx % Cpad);
  float v = 0.0f;
  if (e < Cout) {
    float mean = meanvar[e], var = meanvar[Cpad + e];
    float x = (Y[idx] - mean) * rsqrtf(var + 1e-5f);
    v = fmaxf(gamma[e] * x + beta[e], 0.0f);
  }
  out[idx] = (_Float16)v;
}

// ---------------------------------------------------------------------------
// 9) Max over K + scatter into next-stage pts (B,S,Ctot) and output (B,Ctot,S)
// ---------------------------------------------------------------------------
__global__ void k_maxpool(const _Float16* __restrict__ act, int S, int K,
                          int Cpad, int Cout, int Ctot, int cbase,
                          float* __restrict__ pts_next,
                          float* __restrict__ out_slice) {
  int idx = blockIdx.x * blockDim.x + threadIdx.x;
  if (idx >= 8 * S * Cout) return;
  int e = idx % Cout, s = (idx / Cout) % S, b = idx / (S * Cout);
  const _Float16* a = act + ((size_t)(b * S + s) * K) * Cpad + e;
  float m = -1e30f;
  for (int k = 0; k < K; k++) m = fmaxf(m, (float)a[(size_t)k * Cpad]);
  pts_next[((size_t)b * S + s) * Ctot + cbase + e] = m;
  out_slice[((size_t)b * Ctot + cbase + e) * S + s] = m;
}

// ---------------------------------------------------------------------------
// Host orchestration
// ---------------------------------------------------------------------------
static const int   S_ARR[4]      = {1024, 256, 64, 16};
static const float RAD[4][2]     = {{0.05f, 0.1f}, {0.1f, 0.2f}, {0.2f, 0.4f}, {0.4f, 0.8f}};
static const int   KARR[2]       = {16, 32};
static const int   MLP[4][2][3]  = {{{16, 16, 32},   {32, 32, 64}},
                                    {{64, 64, 128},  {64, 96, 128}},
                                    {{128, 196, 256},{128, 196, 256}},
                                    {{256, 256, 512},{256, 384, 512}}};
static const int   CTOT[4]       = {96, 256, 512, 1024};
static const size_t OUT_OFF[4]   = {0, 786432, 1310720, 1572864};

extern "C" void kernel_launch(void* const* d_in, const int* in_sizes, int n_in,
                              void* d_out, int out_size, void* d_ws, size_t ws_size,
                              hipStream_t stream) {
  (void)out_size; (void)ws_size;
  // xyz may be first or last depending on input flattening order; detect by size.
  int xi = (in_sizes[0] == 8 * 6 * 16384) ? 0 : (n_in - 1);
  int pbase = (xi == 0) ? 1 : 0;
  const float* xyz = (const float*)d_in[xi];
  auto P = [&](int st, int br, int ly, int which) -> const float* {
    // pytree dict order per layer: W(0), b(1), beta(2), gamma(3)
    return (const float*)d_in[pbase + (((st * 2 + br) * 3 + ly) * 4 + which)];
  };

  // Workspace carve-up (all buffers fully rewritten every call).
  size_t off = 0;
  auto alloc = [&](size_t bytes) -> void* {
    off = (off + 255) & ~(size_t)255;
    void* p = (char*)d_ws + off;
    off += bytes;
    return p;
  };
  float* xyz_t  = (float*)alloc((size_t)8 * 16384 * 3 * 4);
  float* pts_t0 = (float*)alloc((size_t)8 * 16384 * 6 * 4);
  float* dist   = (float*)alloc((size_t)8 * 16384 * 4);
  int*   fidx   = (int*)  alloc((size_t)8 * 1024 * 4);
  int*   gidx   = (int*)  alloc((size_t)8 * 1024 * 32 * 4);
  float* newxyz[4]; float* ptsbuf[4];
  for (int st = 0; st < 4; st++) {
    newxyz[st] = (float*)alloc((size_t)8 * S_ARR[st] * 3 * 4);
    ptsbuf[st] = (float*)alloc((size_t)8 * S_ARR[st] * CTOT[st] * 4);
  }
  const size_t XH_MAX  = 8388608;   // max M*Kp   (262144*32 / 65536*128)
  const size_t ACT_MAX = 16777216;  // max M*Cpad (262144*64)
  _Float16* Xh   = (_Float16*)alloc(XH_MAX * 2);
  _Float16* actA = (_Float16*)alloc(ACT_MAX * 2);
  _Float16* actB = (_Float16*)alloc(ACT_MAX * 2);
  float*    Yb   = (float*)   alloc(ACT_MAX * 4);
  _Float16* Wp   = (_Float16*)alloc((size_t)512 * 544 * 2);
  float*    mv   = (float*)   alloc((size_t)2 * 544 * 4);

  k_transpose_in<<<CDIV(8 * 16384, 256), 256, 0, stream>>>(xyz, xyz_t, pts_t0, 16384);

  const float* cur_xyz = xyz_t;
  const float* cur_pts = pts_t0;
  int N = 16384, Cin = 6;

  for (int st = 0; st < 4; st++) {
    int S = S_ARR[st];
    k_fps<<<8, 256, 0, stream>>>(cur_xyz, N, S, dist, fidx, newxyz[st]);
    int cbase = 0;
    for (int br = 0; br < 2; br++) {
      int K = KARR[br];
      float r = RAD[st][br];
      k_ballquery<<<CDIV(8 * S, 256), 256, 0, stream>>>(
          cur_xyz, newxyz[st], N, S, r * r, K, gidx);
      int Kp = RUP32(Cin + 3);
      int M = 8 * S * K;
      k_group<<<CDIV(M, 256), 256, 0, stream>>>(
          cur_pts, cur_xyz, newxyz[st], gidx, N, S, K, Cin, Kp, Xh);

      const _Float16* inact = Xh;
      int inKp = Kp, last = Cin + 3;
      _Float16* finalAct = nullptr;
      int finalCpad = 0, finalCout = 0;
      for (int ly = 0; ly < 3; ly++) {
        int Cout = MLP[st][br][ly];
        int Cpad = RUP32(Cout);
        k_wconv<<<CDIV(Cpad * inKp, 256), 256, 0, stream>>>(
            P(st, br, ly, 0), Cout, last, Cpad, inKp, Wp);
        int waves = (M / 32) * (Cpad / 32);   // 32x32 tile per wave
        k_gemm_wmma<<<CDIV(waves * 32, 256), 256, 0, stream>>>(
            inact, Wp, P(st, br, ly, 1), M, Cpad, inKp, Cout, Yb);
        k_stats<<<Cpad, 256, 0, stream>>>(Yb, M, Cpad, mv);
        _Float16* outact = (ly & 1) ? actB : actA;
        k_bnrelu<<<(unsigned)CDIV((size_t)M * Cpad, 256), 256, 0, stream>>>(
            Yb, mv, P(st, br, ly, 3), P(st, br, ly, 2), M, Cpad, Cout, outact);
        inact = outact; inKp = Cpad; last = Cout;
        finalAct = outact; finalCpad = Cpad; finalCout = Cout;
      }
      k_maxpool<<<CDIV(8 * S * finalCout, 256), 256, 0, stream>>>(
          finalAct, S, K, finalCpad, finalCout, CTOT[st], cbase,
          ptsbuf[st], (float*)d_out + OUT_OFF[st]);
      cbase += finalCout;
    }
    cur_xyz = newxyz[st];
    cur_pts = ptsbuf[st];
    N = S;
    Cin = CTOT[st];
  }
}
